// GAE_75488345195248
// MI455X (gfx1250) — compile-verified, matrix operand order checked by reading, not run on hardware
//
#include <hip/hip_runtime.h>

typedef __attribute__((ext_vector_type(16))) __bf16 v16bf;
typedef __attribute__((ext_vector_type(8)))  __bf16 v8bf;
typedef __attribute__((ext_vector_type(8)))  float  v8f;

#define N_NODES 50000
#define N_EDGES 800000
#define M_TILES 3125          // N_NODES / 16
#define E_TILES 50000         // N_EDGES / 16

// ---------- helpers ----------

__device__ __forceinline__ __bf16 f2bf(float f) {
    union { float f; unsigned u; } in; in.f = f;
    unsigned u = in.u;
    u += 0x7FFFu + ((u >> 16) & 1u);           // round-to-nearest-even
    union { unsigned short s; __bf16 b; } out;
    out.s = (unsigned short)(u >> 16);
    return out.b;
}

// Build a 16-bit A fragment (16x32) from a row-major bf16 row.
// lane 0-15: rows M=0..15, K groups [kbase..kbase+7] and [kbase+16..kbase+23]
// lane 16-31: same rows, K groups shifted by +8.
__device__ __forceinline__ v16bf load_afrag(const __bf16* rowbase, int hi, int kbase) {
    v8bf lo = *(const v8bf*)(rowbase + kbase + hi * 8);
    v8bf hh = *(const v8bf*)(rowbase + kbase + 16 + hi * 8);
    v16bf r;
#pragma unroll
    for (int j = 0; j < 8; ++j) { r[j] = lo[j]; r[j + 8] = hh[j]; }
    return r;
}

__device__ __forceinline__ v8f wmma_bf16(v16bf a, v16bf b, v8f c) {
    return __builtin_amdgcn_wmma_f32_16x16x32_bf16(false, a, false, b, (short)0, c,
                                                   false, false);
}

// ---------- weight prep: pack into B-fragment layout ----------
// frag[(kt*NT+nt)*32 + lane)*16 + j], lane&15 = n within tile, lane>>4 selects K half.

// Combined first-layer weight: Wc[k][n] (64x128):
//   n <  64: Wa[k][n] - Wa[64+k][n]      (applied to x_dst)
//   n >= 64: Wa[64+k][n-64]              (applied to x_src)
__global__ void k_prep_wc(const float* __restrict__ Wa, __bf16* __restrict__ frag) {
    int idx = blockIdx.x * blockDim.x + threadIdx.x;   // 64*128
    if (idx >= 64 * 128) return;
    int k = idx >> 7, n = idx & 127;
    float v = (n < 64) ? (Wa[k * 64 + n] - Wa[(64 + k) * 64 + n])
                       : Wa[(64 + k) * 64 + (n - 64)];
    int kt = k >> 5, kk = k & 31;
    int hi = (kk >> 3) & 1;
    int j  = (kk & 7) + ((kk >= 16) ? 8 : 0);
    int nt = n >> 4;
    int lane = (n & 15) + hi * 16;
    frag[(((size_t)kt * 8 + nt) * 32 + lane) * 16 + j] = f2bf(v);
}

// Second-MLP weight Wb (64x64 row-major, K x N)
__global__ void k_prep_wb(const float* __restrict__ Wb, __bf16* __restrict__ frag) {
    int idx = blockIdx.x * blockDim.x + threadIdx.x;   // 64*64
    if (idx >= 64 * 64) return;
    int k = idx >> 6, n = idx & 63;
    float v = Wb[k * 64 + n];
    int kt = k >> 5, kk = k & 31;
    int hi = (kk >> 3) & 1;
    int j  = (kk & 7) + ((kk >= 16) ? 8 : 0);
    int nt = n >> 4;
    int lane = (n & 15) + hi * 16;
    frag[(((size_t)kt * 4 + nt) * 32 + lane) * 16 + j] = f2bf(v);
}

// ---------- elementwise utility kernels ----------

__global__ void k_fill(float* __restrict__ p, float v, long n) {
    long i = (long)blockIdx.x * blockDim.x + threadIdx.x;
    long stride = (long)gridDim.x * blockDim.x;
    for (; i < n; i += stride) p[i] = v;
}

__global__ void k_cvt_bf16(const float* __restrict__ s, __bf16* __restrict__ d, long n) {
    long i = (long)blockIdx.x * blockDim.x + threadIdx.x;
    long stride = (long)gridDim.x * blockDim.x;
    for (; i < n; i += stride) d[i] = f2bf(s[i]);
}

// ---------- node GEMM: PQ[N x 128] = Xbf[N x 64] @ Wc[64 x 128] (bf16 WMMA) ----------
__global__ void k_node_gemm(const __bf16* __restrict__ Xbf, const __bf16* __restrict__ WcF,
                            float* __restrict__ PQ) {
    int wave = threadIdx.x >> 5;
    int lane = threadIdx.x & 31;
    int mt = blockIdx.x * 4 + wave;
    if (mt >= M_TILES) return;
    int row = lane & 15;
    int hi  = lane >> 4;
    const __bf16* arow = Xbf + (size_t)(mt * 16 + row) * 64;
    v16bf a0 = load_afrag(arow, hi, 0);
    v16bf a1 = load_afrag(arow, hi, 32);
#pragma unroll
    for (int nt = 0; nt < 8; ++nt) {
        v16bf b0 = *(const v16bf*)(WcF + (((size_t)0 * 8 + nt) * 32 + lane) * 16);
        v16bf b1 = *(const v16bf*)(WcF + (((size_t)1 * 8 + nt) * 32 + lane) * 16);
        v8f acc = {0.f, 0.f, 0.f, 0.f, 0.f, 0.f, 0.f, 0.f};
        acc = wmma_bf16(a0, b0, acc);
        acc = wmma_bf16(a1, b1, acc);
        float* out = PQ + (size_t)(mt * 16) * 128 + nt * 16 + (lane & 15);
#pragma unroll
        for (int r = 0; r < 8; ++r) {
            int M = hi ? (r + 8) : r;
            out[(size_t)M * 128] = acc[r];
        }
    }
}

// ---------- edge kernel ----------
// Per 16-edge tile (one wave): rows = ReLU(P[dst]+Q[src]+ba) (bf16 in LDS),
// then [16x64]@Wb[64x64] via WMMA, ReLU, atomic-max scatter into hout[dst].
__global__ void k_edge(const int* __restrict__ src, const int* __restrict__ dst,
                       const float* __restrict__ PQ, const float* __restrict__ ba,
                       const __bf16* __restrict__ WbF, const float* __restrict__ bb,
                       float* __restrict__ hout) {
    __shared__ __align__(32) __bf16 smA[4][16][64];
    __shared__ int sdst[4][16];

    int wave = threadIdx.x >> 5;
    int lane = threadIdx.x & 31;
    int tile = blockIdx.x * 4 + wave;
    int t = tile < E_TILES ? tile : E_TILES - 1;   // clamp keeps EXEC full; dup tiles are max-idempotent
    int e0 = t * 16;

    int er  = e0 + (lane & 15);
    int s_i = src[er];
    int d_i = dst[er];
    if (lane < 16) sdst[wave][lane] = d_i;

    int hi = lane >> 4;
    int cbase = hi * 32;
    const float* prow = PQ + (size_t)d_i * 128 + cbase;         // P part (dst)
    const float* qrow = PQ + (size_t)s_i * 128 + 64 + cbase;    // Q part (src)
    const float* brow = ba + cbase;
#pragma unroll
    for (int c = 0; c < 32; c += 8) {
        v8bf pack;
#pragma unroll
        for (int j = 0; j < 8; ++j) {
            float v = prow[c + j] + qrow[c + j] + brow[c + j];
            pack[j] = f2bf(v > 0.f ? v : 0.f);
        }
        *(v8bf*)&smA[wave][lane & 15][cbase + c] = pack;
    }
    __syncthreads();

    const __bf16* arow = &smA[wave][lane & 15][0];
    v16bf a0 = load_afrag(arow, hi, 0);
    v16bf a1 = load_afrag(arow, hi, 32);

#pragma unroll
    for (int nt = 0; nt < 4; ++nt) {
        v16bf b0 = *(const v16bf*)(WbF + (((size_t)0 * 4 + nt) * 32 + lane) * 16);
        v16bf b1 = *(const v16bf*)(WbF + (((size_t)1 * 4 + nt) * 32 + lane) * 16);
        float cb = bb[nt * 16 + (lane & 15)];
        v8f acc = {cb, cb, cb, cb, cb, cb, cb, cb};
        acc = wmma_bf16(a0, b0, acc);
        acc = wmma_bf16(a1, b1, acc);
#pragma unroll
        for (int r = 0; r < 8; ++r) {
            int M = hi ? (r + 8) : r;
            float v = acc[r] > 0.f ? acc[r] : 0.f;
            float* p = hout + (size_t)sdst[wave][M] * 64 + nt * 16 + (lane & 15);
            asm volatile("global_atomic_max_num_f32 %0, %1, off scope:SCOPE_DEV"
                         :: "v"(p), "v"(v) : "memory");
        }
    }
}

// ---------- BatchNorm ----------
// Pass 1: fix -inf -> 0 (nodes with no incoming edge) and accumulate sum / sumsq.
__global__ void k_bn_stats(float* __restrict__ h, float* __restrict__ sums, long n) {
    __shared__ float red[128];
    int tid = threadIdx.x;
    if (tid < 128) red[tid] = 0.f;
    __syncthreads();
    long i = (long)blockIdx.x * blockDim.x + tid;
    long stride = (long)gridDim.x * blockDim.x;   // multiple of 64 -> channel fixed per thread
    int c = tid & 63;
    float ls = 0.f, lss = 0.f;
    for (; i < n; i += stride) {
        float v = h[i];
        if (v == -__builtin_inff()) { v = 0.f; h[i] = 0.f; }
        ls += v; lss += v * v;
    }
    atomicAdd(&red[c], ls);
    atomicAdd(&red[64 + c], lss);
    __syncthreads();
    if (tid < 128) atomicAdd(&sums[tid], red[tid]);
}

__global__ void k_bn_finalize(float* __restrict__ sums, const float* __restrict__ gamma,
                              const float* __restrict__ beta, float n) {
    int c = threadIdx.x;
    if (c < 64) {
        float mean = sums[c] / n;
        float var  = sums[64 + c] / n - mean * mean;
        float sc   = gamma[c] * rsqrtf(var + 1e-5f);
        sums[128 + c] = sc;
        sums[192 + c] = beta[c] - mean * sc;
    }
}

__global__ void k_bn_apply(const float* __restrict__ h, const float* __restrict__ sums,
                           float* __restrict__ of32, __bf16* __restrict__ obf, long n) {
    long i = (long)blockIdx.x * blockDim.x + threadIdx.x;
    long stride = (long)gridDim.x * blockDim.x;
    for (; i < n; i += stride) {
        int c = (int)(i & 63);
        float y = h[i] * sums[128 + c] + sums[192 + c];
        if (of32) of32[i] = y;
        if (obf)  obf[i]  = f2bf(y);
    }
}

// ---------- launcher ----------
extern "C" void kernel_launch(void* const* d_in, const int* in_sizes, int n_in,
                              void* d_out, int out_size, void* d_ws, size_t ws_size,
                              hipStream_t stream) {
    (void)in_sizes; (void)n_in; (void)out_size; (void)ws_size;

    const float* x     = (const float*)d_in[0];
    const int*   ei    = (const int*)d_in[1];
    const int*   src   = ei;
    const int*   dst   = ei + N_EDGES;
    const float* W1a   = (const float*)d_in[3];
    const float* b1a   = (const float*)d_in[4];
    const float* W1b   = (const float*)d_in[5];
    const float* b1b   = (const float*)d_in[6];
    const float* g1    = (const float*)d_in[7];
    const float* be1   = (const float*)d_in[8];
    const float* W2a   = (const float*)d_in[9];
    const float* b2a   = (const float*)d_in[10];
    const float* W2b   = (const float*)d_in[11];
    const float* b2b   = (const float*)d_in[12];
    const float* g2    = (const float*)d_in[13];
    const float* be2   = (const float*)d_in[14];

    char* ws = (char*)d_ws;
    __bf16* WcF1 = (__bf16*)(ws + 0);         // 16 KB
    __bf16* WbF1 = (__bf16*)(ws + 16384);     //  8 KB
    __bf16* WcF2 = (__bf16*)(ws + 24576);     // 16 KB
    __bf16* WbF2 = (__bf16*)(ws + 40960);     //  8 KB
    float*  sums = (float*)(ws + 49152);      //  1 KB (sum | sumsq | scale | shift)
    __bf16* Xbf  = (__bf16*)(ws + 50176);                         // 6.4 MB
    float*  PQ   = (float*)(ws + 50176 + (size_t)N_NODES * 64 * 2); // 25.6 MB
    float*  h    = (float*)d_out;             // max-accumulator, reused as final output

    const float NEG_INF = -__builtin_huge_valf();
    const long  NH = (long)N_NODES * 64;

    // weight prep (fragment layout) + input conversion
    k_prep_wc<<<32, 256, 0, stream>>>(W1a, WcF1);
    k_prep_wc<<<32, 256, 0, stream>>>(W2a, WcF2);
    k_prep_wb<<<16, 256, 0, stream>>>(W1b, WbF1);
    k_prep_wb<<<16, 256, 0, stream>>>(W2b, WbF2);
    k_cvt_bf16<<<2048, 256, 0, stream>>>(x, Xbf, NH);

    // ---- layer 1 ----
    k_node_gemm<<<(M_TILES + 3) / 4, 128, 0, stream>>>(Xbf, WcF1, PQ);
    k_fill<<<2048, 256, 0, stream>>>(h, NEG_INF, NH);
    k_edge<<<E_TILES / 4, 128, 0, stream>>>(src, dst, PQ, b1a, WbF1, b1b, h);
    k_fill<<<1, 256, 0, stream>>>(sums, 0.f, 128);
    k_bn_stats<<<1024, 256, 0, stream>>>(h, sums, NH);
    k_bn_finalize<<<1, 64, 0, stream>>>(sums, g1, be1, (float)N_NODES);
    k_bn_apply<<<2048, 256, 0, stream>>>(h, sums, nullptr, Xbf, NH);  // bf16 input for layer 2

    // ---- layer 2 ----
    k_node_gemm<<<(M_TILES + 3) / 4, 128, 0, stream>>>(Xbf, WcF2, PQ);
    k_fill<<<2048, 256, 0, stream>>>(h, NEG_INF, NH);
    k_edge<<<E_TILES / 4, 128, 0, stream>>>(src, dst, PQ, b2a, WbF2, b2b, h);
    k_fill<<<1, 256, 0, stream>>>(sums, 0.f, 128);
    k_bn_stats<<<1024, 256, 0, stream>>>(h, sums, NH);
    k_bn_finalize<<<1, 64, 0, stream>>>(sums, g2, be2, (float)N_NODES);
    k_bn_apply<<<2048, 256, 0, stream>>>(h, sums, h, nullptr, NH);    // in-place f32 output
}
